// MultiHeadAttention_22488448762429
// MI455X (gfx1250) — matrix operand order repaired and checked by reading, not compiled, not run on hardware
//
#include <hip/hip_runtime.h>
#include <hip/hip_bf16.h>
#include <math.h>

#define S_LEN 2048
#define E_DIM 2048
#define DH    128
#define NH    16
#define NKVH  4

typedef __bf16 bf16_t;
typedef __attribute__((ext_vector_type(16))) __bf16 v16bf;
typedef __attribute__((ext_vector_type(8)))  __bf16 v8bf;
typedef __attribute__((ext_vector_type(4)))  __bf16 v4bf;
typedef __attribute__((ext_vector_type(8)))  float  v8f;
typedef __attribute__((ext_vector_type(4)))  float  v4f;
typedef __attribute__((ext_vector_type(4)))  unsigned int u32x4;
typedef __attribute__((ext_vector_type(8)))  int i32x8;
typedef __attribute__((ext_vector_type(4)))  int i32x4;

// LDS B-panel row stride: 32 K-elements (64B) + 16B TDM pad -> 80B, conflict-free b128 reads
#define BPANEL_ROW_ELEMS 40

static __device__ __forceinline__ bf16_t f2bf(float f) {
  union { float f; unsigned u; } v; v.f = f;
  unsigned r = v.u + 0x7FFFu + ((v.u >> 16) & 1u);   // round-to-nearest-even
  unsigned short h = (unsigned short)(r >> 16);
  return __builtin_bit_cast(bf16_t, h);
}

static __device__ __forceinline__ v8f vzero8() {
  v8f z;
#pragma unroll
  for (int i = 0; i < 8; ++i) z[i] = 0.0f;
  return z;
}

static __device__ __forceinline__ v8f wmma_bf16(v16bf a, v16bf b, v8f c) {
  return __builtin_amdgcn_wmma_f32_16x16x32_bf16(false, a, false, b, (short)0, c, false, false);
}

// A fragment: 16(M) x 32(K) bf16 from row-major [.., lda] source.
static __device__ __forceinline__ v16bf load_a(const bf16_t* __restrict__ A, int lda,
                                               int row0, int k0, int lane) {
  int r  = row0 + (lane & 15);
  int kb = k0 + ((lane >> 4) << 3);
  const bf16_t* p = A + (size_t)r * lda + kb;
  v8bf lo = *(const v8bf*)(p);
  v8bf hi = *(const v8bf*)(p + 16);
  v16bf o;
#pragma unroll
  for (int i = 0; i < 8; ++i) { o[i] = lo[i]; o[i + 8] = hi[i]; }
  return o;
}

// B fragment from global: 32(K) x 16(N); column n == contiguous-K row (col0+n) of W.
static __device__ __forceinline__ v16bf load_b(const bf16_t* __restrict__ W, int ldw,
                                               int col0, int k0, int lane) {
  int n  = col0 + (lane & 15);
  int kb = k0 + ((lane >> 4) << 3);
  const bf16_t* p = W + (size_t)n * ldw + kb;
  v8bf lo = *(const v8bf*)(p);
  v8bf hi = *(const v8bf*)(p + 16);
  v16bf o;
#pragma unroll
  for (int i = 0; i < 8; ++i) { o[i] = lo[i]; o[i + 8] = hi[i]; }
  return o;
}

// B fragment from an LDS panel [128 rows][40 elems] (TDM-padded rows).
static __device__ __forceinline__ v16bf load_b_lds(const bf16_t* sbuf, int t, int lane) {
  const bf16_t* p = sbuf + (t * 16 + (lane & 15)) * BPANEL_ROW_ELEMS + ((lane >> 4) << 3);
  v8bf lo = *(const v8bf*)(p);
  v8bf hi = *(const v8bf*)(p + 16);
  v16bf o;
#pragma unroll
  for (int i = 0; i < 8; ++i) { o[i] = lo[i]; o[i + 8] = hi[i]; }
  return o;
}

// TDM: DMA a 128-row x 32-element bf16 B panel (rows = weight rows, contiguous K)
// from global into LDS, padding 16B after each 64B row (bank-conflict-free reads).
// D# packing per CDNA5 ISA. 6-arg builtin form
// (clang-23 / therock-10.0): (u32x4 g0, i32x8 g1, i32x4 g2, i32x4 g3, i32x8 gx, i32 cpol).
static __device__ __forceinline__ void tdm_load_panel(const void* gptr, unsigned lds_off) {
  unsigned long long ga = (unsigned long long)(size_t)gptr;
  u32x4 g0;
  g0[0] = 1u;                                              // count=1 (valid user D#)
  g0[1] = lds_off;                                         // lds_addr (bytes)
  g0[2] = (unsigned)ga;                                    // global_addr[31:0]
  g0[3] = (unsigned)((ga >> 32) & 0x01FFFFFFu)             // global_addr[56:32]
        | 0x80000000u;                                     // type=2 ("image")
  i32x8 g1;
  g1[0] = (int)0x06D10000;   // data_size=2B | pad_enable | pad_interval=16DW | pad_amount=4DW
  g1[1] = (int)0x08000000;   // tensor_dim0 = 2048 -> bits[79:48] low half
  g1[2] = (int)0x00800000;   // tensor_dim1 = 128  -> bits[111:80] low half
  g1[3] = (int)0x00200000;   // tensor_dim1 hi=0, tile_dim0 = 32 -> bits[127:112]
  g1[4] = 128;               // tile_dim1  = 128
  g1[5] = 2048;              // tensor_dim0_stride = 2048 (low 32 of 48)
  g1[6] = 0;
  g1[7] = 0;                 // tensor_dim1_stride unused (2D tile)
  i32x4 z4 = {0, 0, 0, 0};
  i32x8 z8;
#pragma unroll
  for (int i = 0; i < 8; ++i) z8[i] = 0;
  __builtin_amdgcn_tensor_load_to_lds(g0, g1, z4, z4, z8, 0);
}

static __device__ __forceinline__ unsigned lds_off_of(const void* p) {
  return (unsigned)(size_t)p;   // flat LDS address: offset lives in addr[31:0]
}

// ---------------- fp32 -> bf16 conversion ----------------
__global__ __launch_bounds__(256) void cvt_f32_bf16(const float* __restrict__ src,
                                                    bf16_t* __restrict__ dst, int n4) {
  int i = blockIdx.x * blockDim.x + threadIdx.x;
  int stride = gridDim.x * blockDim.x;
  for (; i < n4; i += stride) {
    v4f v = ((const v4f*)src)[i];
    v4bf o;
#pragma unroll
    for (int j = 0; j < 4; ++j) o[j] = f2bf(v[j]);
    ((v4bf*)dst)[i] = o;
  }
}

// ---------------- QKV projection + RMSNorm + RoPE ----------------
// 4 waves/block share one head's B panel (TDM double-buffered in LDS);
// each wave owns a different 16-row block.
__global__ __launch_bounds__(128) void qkv_gemm(
    const bf16_t* __restrict__ xb,  const bf16_t* __restrict__ wqb,
    const bf16_t* __restrict__ wkb, const bf16_t* __restrict__ wvb,
    const float*  __restrict__ qnw, const float*  __restrict__ knw,
    bf16_t* __restrict__ Qh, bf16_t* __restrict__ Kh, bf16_t* __restrict__ VTh) {
  __shared__ bf16_t sB[2][128 * BPANEL_ROW_ELEMS];
  const int lane  = threadIdx.x & 31;
  const int wid   = threadIdx.x >> 5;
  const int strip = blockIdx.x * 4 + wid;   // 0..3071

  int seg, rb, hd;
  const bf16_t* W;
  if (strip < 2048)      { seg = 0; hd = strip >> 7;          rb = strip & 127;          W = wqb; }
  else if (strip < 2560) { seg = 1; hd = (strip - 2048) >> 7; rb = (strip - 2048) & 127; W = wkb; }
  else                   { seg = 2; hd = (strip - 2560) >> 7; rb = (strip - 2560) & 127; W = wvb; }
  const int col0 = hd * DH;            // block-uniform (4 waves share hd)
  const int row0 = rb * 16;

  v8f acc[8];
#pragma unroll
  for (int t = 0; t < 8; ++t) acc[t] = vzero8();

  const int NSTEP = E_DIM / 32;        // 64
  if (wid == 0) tdm_load_panel(W + (size_t)col0 * E_DIM, lds_off_of(&sB[0][0]));

  for (int step = 0; step < NSTEP; ++step) {
    if (wid == 0) {
      if (step + 1 < NSTEP) {
        tdm_load_panel(W + (size_t)col0 * E_DIM + (step + 1) * 32,
                       lds_off_of(&sB[(step + 1) & 1][0]));
        __builtin_amdgcn_s_wait_tensorcnt(1);   // current panel has landed
      } else {
        __builtin_amdgcn_s_wait_tensorcnt(0);
      }
    }
    __syncthreads();                            // panel visible to all waves
    v16bf a = load_a(xb, E_DIM, row0, step * 32, lane);
    const bf16_t* sb = sB[step & 1];
    // Issue all 16 ds_load_b128 first, then drain with WMMAs so LDS latency
    // is covered by 7 outstanding fragment loads instead of 1.
    v16bf bfr[8];
#pragma unroll
    for (int t = 0; t < 8; ++t) bfr[t] = load_b_lds(sb, t, lane);
#pragma unroll
    for (int t = 0; t < 8; ++t) acc[t] = wmma_bf16(a, bfr[t], acc[t]);
    __syncthreads();                            // all reads done before buffer reuse
  }

  const int hi8 = (lane >> 4) << 3;             // C layout: VGPR r holds row (r + hi8)
  if (seg == 2) {
    // V stored transposed VT[kvh][d][s] so PV-WMMA B loads are contiguous in keys.
    bf16_t* base = VTh + (size_t)hd * DH * S_LEN;
#pragma unroll
    for (int t = 0; t < 8; ++t) {
      int d = t * 16 + (lane & 15);
#pragma unroll
      for (int r = 0; r < 8; ++r) {
        int row = row0 + r + hi8;
        base[(size_t)d * S_LEN + row] = f2bf(acc[t][r]);
      }
    }
    return;
  }

  // ---- QK-RMSNorm over Dh=128 ----
  const float* nw = (seg == 0) ? qnw : knw;
  float wv_[8];
#pragma unroll
  for (int t = 0; t < 8; ++t) wv_[t] = nw[t * 16 + (lane & 15)];
  float rms[8];
#pragma unroll
  for (int r = 0; r < 8; ++r) {
    float s = 0.0f;
#pragma unroll
    for (int t = 0; t < 8; ++t) s += acc[t][r] * acc[t][r];
#pragma unroll
    for (int m = 1; m < 16; m <<= 1) s += __shfl_xor(s, m, 32);
    rms[r] = rsqrtf(s * (1.0f / 128.0f) + 1e-6f);
  }
#pragma unroll
  for (int t = 0; t < 8; ++t)
#pragma unroll
    for (int r = 0; r < 8; ++r) acc[t][r] *= rms[r] * wv_[t];

  // ---- RoPE: d and d+64 live in tiles t and t+4 of the same lane ----
#pragma unroll
  for (int t = 0; t < 4; ++t) {
    int d = t * 16 + (lane & 15);
    float invf = __expf(-(float)d * (9.210340371976184f / 64.0f)); // 10000^(-d/64)
#pragma unroll
    for (int r = 0; r < 8; ++r) {
      int row = row0 + r + hi8;
      float sn, cs;
      __sincosf((float)row * invf, &sn, &cs);
      float lo = acc[t][r], hi = acc[t + 4][r];
      acc[t][r]     = lo * cs - hi * sn;
      acc[t + 4][r] = hi * cs + lo * sn;
    }
  }

  bf16_t* base = (seg == 0 ? Qh : Kh) + (size_t)hd * S_LEN * DH;
#pragma unroll
  for (int t = 0; t < 8; ++t) {
    int d = t * 16 + (lane & 15);
#pragma unroll
    for (int r = 0; r < 8; ++r) {
      int row = row0 + r + hi8;
      base[(size_t)row * DH + d] = f2bf(acc[t][r]);
    }
  }
}

// ---------------- causal flash attention ----------------
__global__ __launch_bounds__(128) void attn_fa(
    const bf16_t* __restrict__ Qh, const bf16_t* __restrict__ Kh,
    const bf16_t* __restrict__ VTh, bf16_t* __restrict__ AO) {
  __shared__ float sP[4][16][33];   // per-wave P staging (C-layout -> A-layout)
  const int lane = threadIdx.x & 31;
  const int w    = threadIdx.x >> 5;
  const int gw   = blockIdx.x * 4 + w;     // 0..2047
  const int head = gw >> 7;
  const int qb   = gw & 127;
  const int kvh  = head >> 2;              // repeat_interleave(rep=4)
  const int hi8  = (lane >> 4) << 3;

  const bf16_t* Qbase = Qh  + (size_t)head * S_LEN * DH;
  const bf16_t* Kbase = Kh  + (size_t)kvh  * S_LEN * DH;
  const bf16_t* Vbase = VTh + (size_t)kvh  * DH * S_LEN;

  v16bf qf[4];
#pragma unroll
  for (int c = 0; c < 4; ++c) qf[c] = load_a(Qbase, DH, qb * 16, c * 32, lane);

  v8f o[8];
#pragma unroll
  for (int t = 0; t < 8; ++t) o[t] = vzero8();
  float mrun[8], lrun[8];
#pragma unroll
  for (int r = 0; r < 8; ++r) { mrun[r] = -3.0e38f; lrun[r] = 0.0f; }

  const float scale = 0.08838834764831845f;  // 1/sqrt(128)
  const int nkb = ((qb * 16 + 16) + 31) >> 5;

  for (int blk = 0; blk < nkb; ++blk) {
    const int key0 = blk * 32;
    v8f st[2];
#pragma unroll
    for (int tn = 0; tn < 2; ++tn) {
      v8f s = vzero8();
#pragma unroll
      for (int c = 0; c < 4; ++c) {
        v16bf bf = load_b(Kbase, DH, key0 + tn * 16, c * 32, lane);
        s = wmma_bf16(qf[c], bf, s);
      }
      int key = key0 + tn * 16 + (lane & 15);
#pragma unroll
      for (int r = 0; r < 8; ++r) {
        int q = qb * 16 + r + hi8;
        st[tn][r] = (key > q) ? -1.0e30f : (s[r] * scale);
      }
    }
    // online softmax
#pragma unroll
    for (int r = 0; r < 8; ++r) {
      float bm = fmaxf(st[0][r], st[1][r]);
#pragma unroll
      for (int m = 1; m < 16; m <<= 1) bm = fmaxf(bm, __shfl_xor(bm, m, 32));
      float mnew = fmaxf(mrun[r], bm);
      float corr = __expf(mrun[r] - mnew);
      float p0 = __expf(st[0][r] - mnew);
      float p1 = __expf(st[1][r] - mnew);
      st[0][r] = p0; st[1][r] = p1;
      float ls = p0 + p1;
#pragma unroll
      for (int m = 1; m < 16; m <<= 1) ls += __shfl_xor(ls, m, 32);
      lrun[r] = lrun[r] * corr + ls;
      mrun[r] = mnew;
#pragma unroll
      for (int t = 0; t < 8; ++t) o[t][r] *= corr;
    }
    // P: C-layout -> A-layout through wave-private LDS
#pragma unroll
    for (int tn = 0; tn < 2; ++tn)
#pragma unroll
      for (int r = 0; r < 8; ++r)
        sP[w][r + hi8][tn * 16 + (lane & 15)] = st[tn][r];
    asm volatile("s_wait_dscnt 0" ::: "memory");
    v16bf pa;
    {
      int m = lane & 15;
#pragma unroll
      for (int i = 0; i < 8; ++i) {
        pa[i]     = f2bf(sP[w][m][hi8 + i]);
        pa[i + 8] = f2bf(sP[w][m][16 + hi8 + i]);
      }
    }
    // O += P * V
#pragma unroll
    for (int t = 0; t < 8; ++t) {
      v16bf bf = load_b(Vbase, S_LEN, t * 16, key0, lane);
      o[t] = wmma_bf16(pa, bf, o[t]);
    }
  }

#pragma unroll
  for (int r = 0; r < 8; ++r) {
    float inv = 1.0f / lrun[r];
    int row = qb * 16 + r + hi8;
#pragma unroll
    for (int t = 0; t < 8; ++t) {
      int col = head * DH + t * 16 + (lane & 15);
      AO[(size_t)row * E_DIM + col] = f2bf(o[t][r] * inv);
    }
  }
}

// ---------------- output projection: out = AO @ Wo^T (f32 out) ----------------
// Same TDM-staged B-panel pipeline as qkv_gemm.
__global__ __launch_bounds__(128) void out_proj(
    const bf16_t* __restrict__ AO, const bf16_t* __restrict__ wob,
    float* __restrict__ out) {
  __shared__ bf16_t sB[2][128 * BPANEL_ROW_ELEMS];
  const int lane  = threadIdx.x & 31;
  const int wid   = threadIdx.x >> 5;
  const int strip = blockIdx.x * 4 + wid;  // 0..2047
  const int cb = strip >> 7;               // block-uniform column strip
  const int rb = strip & 127;
  const int col0 = cb * 128;
  const int row0 = rb * 16;

  v8f acc[8];
#pragma unroll
  for (int t = 0; t < 8; ++t) acc[t] = vzero8();

  const int NSTEP = E_DIM / 32;
  if (wid == 0) tdm_load_panel(wob + (size_t)col0 * E_DIM, lds_off_of(&sB[0][0]));

  for (int step = 0; step < NSTEP; ++step) {
    if (wid == 0) {
      if (step + 1 < NSTEP) {
        tdm_load_panel(wob + (size_t)col0 * E_DIM + (step + 1) * 32,
                       lds_off_of(&sB[(step + 1) & 1][0]));
        __builtin_amdgcn_s_wait_tensorcnt(1);
      } else {
        __builtin_amdgcn_s_wait_tensorcnt(0);
      }
    }
    __syncthreads();
    v16bf a = load_a(AO, E_DIM, row0, step * 32, lane);
    const bf16_t* sb = sB[step & 1];
    v16bf bfr[8];
#pragma unroll
    for (int t = 0; t < 8; ++t) bfr[t] = load_b_lds(sb, t, lane);
#pragma unroll
    for (int t = 0; t < 8; ++t) acc[t] = wmma_bf16(a, bfr[t], acc[t]);
    __syncthreads();
  }

  const int hi8 = (lane >> 4) << 3;
#pragma unroll
  for (int t = 0; t < 8; ++t) {
    int col = col0 + t * 16 + (lane & 15);
#pragma unroll
    for (int r = 0; r < 8; ++r) {
      int row = rb * 16 + r + hi8;
      out[(size_t)row * E_DIM + col] = acc[t][r];
    }
  }
}

extern "C" void kernel_launch(void* const* d_in, const int* in_sizes, int n_in,
                              void* d_out, int out_size, void* d_ws, size_t ws_size,
                              hipStream_t stream) {
  const float* x   = (const float*)d_in[0];
  // d_in[1] = causal mask (bool) — structure known, unused
  const float* Wq  = (const float*)d_in[2];
  const float* Wk  = (const float*)d_in[3];
  const float* Wv  = (const float*)d_in[4];
  const float* Wo  = (const float*)d_in[5];
  const float* qnw = (const float*)d_in[6];
  const float* knw = (const float*)d_in[7];

  bf16_t* p = (bf16_t*)d_ws;
  bf16_t* xb  = p; p += (size_t)S_LEN * E_DIM;
  bf16_t* wqb = p; p += (size_t)E_DIM * E_DIM;
  bf16_t* wkb = p; p += (size_t)NKVH * DH * E_DIM;
  bf16_t* wvb = p; p += (size_t)NKVH * DH * E_DIM;
  bf16_t* wob = p; p += (size_t)E_DIM * E_DIM;
  bf16_t* Qh  = p; p += (size_t)NH * S_LEN * DH;
  bf16_t* Kh  = p; p += (size_t)NKVH * S_LEN * DH;
  bf16_t* VTh = p; p += (size_t)NKVH * DH * S_LEN;
  bf16_t* AO  = p; p += (size_t)S_LEN * E_DIM;

  auto cvt = [&](const float* s, bf16_t* d, int n) {
    int n4 = n / 4;
    int blocks = (n4 + 255) / 256;
    cvt_f32_bf16<<<blocks, 256, 0, stream>>>(s, d, n4);
  };
  cvt(x,  xb,  S_LEN * E_DIM);
  cvt(Wq, wqb, E_DIM * E_DIM);
  cvt(Wk, wkb, NKVH * DH * E_DIM);
  cvt(Wv, wvb, NKVH * DH * E_DIM);
  cvt(Wo, wob, E_DIM * E_DIM);

  qkv_gemm<<<768, 128, 0, stream>>>(xb, wqb, wkb, wvb, qnw, knw, Qh, Kh, VTh);
  attn_fa<<<512, 128, 0, stream>>>(Qh, Kh, VTh, AO);
  out_proj<<<512, 128, 0, stream>>>(AO, wob, (float*)d_out);
}
